// PatchEmbed_47090021433561
// MI455X (gfx1250) — matrix operand order repaired and checked by reading, not compiled, not run on hardware
//
#include <hip/hip_runtime.h>
#include <hip/hip_bf16.h>

// ---------------------------------------------------------------------------
// PatchEmbed (polar) for MI455X / gfx1250, wave32 + WMMA bf16 + TDM.
//   K0: prep   : radius table, weights f32->bf16, theta_max
//   K1: fused bilinear grid-sample + conv1 (K=60->64), W1 fragments hoisted to
//       registers so global loads overlap the sampling math; writes t1 in bf16
//       conv2-patch-major with packed b64 stores.
//   K2: conv2 = GEMM M=32768,K=1920,N=96; double-buffered Tensor-Data-Mover
//       (tensor_load_to_lds) pipeline feeding v_wmma_f32_16x16x32_bf16.
// ---------------------------------------------------------------------------

typedef __attribute__((ext_vector_type(16))) __bf16         bf16x16;
typedef __attribute__((ext_vector_type(8)))  float          f32x8;
typedef __attribute__((ext_vector_type(8)))  unsigned short ushort8;
typedef __attribute__((ext_vector_type(4)))  unsigned short ushort4v;
typedef __attribute__((ext_vector_type(4)))  unsigned int   u32x4;
typedef __attribute__((ext_vector_type(8)))  int            i32x8;
typedef __attribute__((ext_vector_type(4)))  int            i32x4;

#define BATCH     32
#define IMGSZ     224
#define NRAD      400
#define NAZ       1024
#define EMBED     96
#define H1        80     // conv1 out rows
#define KPAD1     64     // conv1 K (60) padded to 64
#define K2DIM     1920   // conv2 K = 96*5*4
#define NTOK      1024   // 16*64 tokens per batch
#define PI_F      3.14159265358979323846f
#define HALF_PI_F 1.57079632679489662f

// ---- workspace layout (bytes) ----
#define WS_RTAB   0                      // 32*400 f32          = 51200 B
#define WS_W1T    51200                  // 96*64  bf16(ushort) = 12288 B
#define WS_W2T    63488                  // 96*1920 ushort      = 368640 B
#define WS_T1     432128                 // 32*1024*1920 ushort = 125829120 B

#if defined(__gfx1250__) && __has_builtin(__builtin_amdgcn_tensor_load_to_lds)
#define HAVE_TDM 1
#else
#define HAVE_TDM 0
#endif

__device__ __forceinline__ unsigned short f32_to_bf16_u(float f) {
    union { float f; unsigned u; } v; v.f = f;
    unsigned r = v.u + 0x7FFFu + ((v.u >> 16) & 1u);   // round to nearest even
    return (unsigned short)(r >> 16);
}

union FragBits { ushort8 h[2]; bf16x16 v; };

// A-matrix fragment (16x32 bf16), ISA 7.12.2 layout:
// lane L: row M = L&15 ; K blocks {0-7,16-23} (L<16) or {8-15,24-31} (L>=16)
__device__ __forceinline__ bf16x16 load_frag_a(const unsigned short* base, int lane, int ldk) {
    int m  = lane & 15;
    int kb = (lane & 16) ? 8 : 0;
    const unsigned short* p = base + m * ldk + kb;
    FragBits fb;
    fb.h[0] = *(const ushort8*)(p);
    fb.h[1] = *(const ushort8*)(p + 16);
    return fb.v;
}

// B-matrix fragment (32x16 bf16): lane L: col N = L&15 ;
// K blocks {0-7,16-23} (L<16) or {8-15,24-31} (L>=16) (Wt is [o][k]).
__device__ __forceinline__ bf16x16 load_frag_b(const unsigned short* base, int lane, int ldk) {
    int n  = lane & 15;
    int kb = (lane & 16) ? 8 : 0;
    const unsigned short* p = base + n * ldk + kb;
    FragBits fb;
    fb.h[0] = *(const ushort8*)(p);
    fb.h[1] = *(const ushort8*)(p + 16);
    return fb.v;
}

#if HAVE_TDM
// Issue one 2-D bf16 tile DMA: global (row_stride_elems pitch) -> LDS, with
// LDS padding of 4 DWORDs after every 16 DWORDs (64B row -> 80B LDS stride).
// D# per CDNA5 ISA 8.3-8.5 (data_size=1 -> 2 bytes, type=2, count=1).
// This toolchain exposes the 6-argument builtin:
//   (u32x4 g0, i32x8 g1, i32x4 g2, i32x4 g3, i32x8 extra, i32 cpol)
__device__ __forceinline__ void tdm_load_tile_2d(
    const void* gsrc, void* ldst,
    unsigned tile_w_elems, unsigned tile_h,
    unsigned tensor_w_elems, unsigned tensor_h,
    unsigned row_stride_elems) {
    unsigned long long ga = (unsigned long long)(size_t)gsrc;
    u32x4 g0;
    g0[0] = 1u;                                        // count=1 (valid), user mode
    g0[1] = (unsigned)(size_t)ldst;                    // lds_addr (low 32 bits of flat)
    g0[2] = (unsigned)ga;                              // global_addr[31:0]
    g0[3] = (unsigned)((ga >> 32) & 0x01FFFFFFull)     // global_addr[56:32]
          | (2u << 30);                                // type=2 ("image")
    const unsigned d0 = (1u << 16)                     // data_size = 2 bytes
                      | (1u << 20)                     // pad_enable
                      | (3u << 22)                     // pad_interval: 16 DWORDs
                      | (3u << 25);                    // pad_amount: 4 DWORDs
    unsigned long long q0 = (unsigned long long)d0
                          | ((unsigned long long)(tensor_w_elems & 0xFFFFu) << 48);
    unsigned long long q1 = (unsigned long long)(tensor_w_elems >> 16)
                          | ((unsigned long long)tensor_h << 16)
                          | ((unsigned long long)(tile_w_elems & 0xFFFFu) << 48);
    unsigned long long s0 = row_stride_elems;          // tensor_dim0_stride (48b)
    unsigned long long q2 = (unsigned long long)(tile_h & 0xFFFFu)
                          | ((s0 & 0xFFFFFFFFull) << 32);
    unsigned long long q3 = (s0 >> 32) & 0xFFFFull;    // dim1_stride=0 (2-D tile)
    union { unsigned long long q[4]; i32x8 v; } g1;
    g1.q[0] = q0; g1.q[1] = q1; g1.q[2] = q2; g1.q[3] = q3;
    i32x4 gz4 = {0, 0, 0, 0};                          // groups 2/3: dims unused
    i32x8 gz8 = {0, 0, 0, 0, 0, 0, 0, 0};
    __builtin_amdgcn_tensor_load_to_lds(g0, g1.v, gz4, gz4, gz8, 0);
}
#endif

// ---------------------------------------------------------------------------
// K0: radius table + bf16 weight repack + theta_max (grid 128 x 256)
// ---------------------------------------------------------------------------
__global__ __launch_bounds__(256) void prep_kernel(
    const float* __restrict__ dist, const float* __restrict__ w1,
    const float* __restrict__ w2, float* __restrict__ r_table,
    unsigned short* __restrict__ W1t, unsigned short* __restrict__ W2t,
    float* __restrict__ theta_out) {
    int blk = blockIdx.x, tid = threadIdx.x;
    if (blk < BATCH) {
        int b = blk;
        float c0 = 0.2f + dist[b * 4 + 0];
        float c1 = 0.2f + dist[b * 4 + 1];
        float c2 = 0.2f + dist[b * 4 + 2];
        float c3 = 0.2f + dist[b * 4 + 3];
        auto poly = [&](float t) {
            float t2 = t * t;
            return t * (c0 + t2 * (c1 + t2 * (c2 + t2 * c3)));
        };
        float scale = 112.0f / poly(HALF_PI_F);     // * (H*0.5)
        for (int i = tid; i < NRAD; i += 256) {
            float th = HALF_PI_F * (i + 0.5f) * (1.0f / NRAD);
            r_table[b * NRAD + i] = poly(th) * scale;
        }
        if (tid == 0) theta_out[b] = HALF_PI_F;
    } else {
        int o = blk - BATCH;                        // 0..95
        // OIHW with stride==kernel: Wt[o][k] == w_flat[o*K+k] (pure cvt)
        for (int k = tid; k < K2DIM; k += 256)
            W2t[o * K2DIM + k] = f32_to_bf16_u(w2[(size_t)o * K2DIM + k]);
        for (int k = tid; k < KPAD1; k += 256)
            W1t[o * KPAD1 + k] = (k < 60) ? f32_to_bf16_u(w1[o * 60 + k])
                                          : (unsigned short)0;
    }
}

// ---------------------------------------------------------------------------
// K1: fused bilinear sample + conv1.
// One block = (b, ro, 64 consecutive ao). 128 threads = 4 waves.
// ---------------------------------------------------------------------------
__global__ __launch_bounds__(128) void sample_conv1_kernel(
    const float* __restrict__ x, const float* __restrict__ r_table,
    const unsigned short* __restrict__ W1t, const float* __restrict__ b1,
    unsigned short* __restrict__ T1) {
    __shared__ unsigned short Atile[64 * 72];       // 64 rows, K stride 72 (16B-aligned)
    const int tid = threadIdx.x;
    const int wg  = blockIdx.x;
    const int b   = wg / 320;
    const int rem = wg % 320;
    const int ro  = rem >> 2;
    const int ao0 = (rem & 3) * 64;
    const int lane = tid & 31, w = tid >> 5;

    // Hoist all 12 loop-invariant W1 B-fragments into registers NOW so the
    // global loads overlap the gather/transcendental phase below.
    bf16x16 bw[12];
    #pragma unroll
    for (int kk = 0; kk < 2; ++kk)
        #pragma unroll
        for (int j = 0; j < 6; ++j)
            bw[kk * 6 + j] = load_frag_b(W1t + (j * 16) * KPAD1 + kk * 32, lane, KPAD1);

    // zero K padding (k = 60..63)
    for (int i = tid; i < 64 * 4; i += 128)
        Atile[(i >> 2) * 72 + 60 + (i & 3)] = 0;

    // gather phase: 64 spatial x 20 taps, 3 channels each
    for (int p = tid; p < 64 * 20; p += 128) {
        int spos = p / 20;
        int q    = p - spos * 20;                   // kr*4 + ka
        int kr   = q >> 2, ka = q & 3;
        int r_idx = ro * 5 + kr;                    // 0..399
        int a_idx = (ao0 + spos) * 4 + ka;          // 0..1023
        float radius = r_table[b * NRAD + r_idx];
        float phi = 2.0f * PI_F * (a_idx + 0.5f) * (1.0f / NAZ);
        float sp = sinf(phi), cp = cosf(phi);
        float xc = radius * cp, yc = radius * sp;
        // grid[...,0] = yc/112 -> gx ; grid[...,1] = xc/112 -> gy (ref swaps)
        float gx = (yc * (1.0f / 112.0f) + 1.0f) * 0.5f * 223.0f;
        float gy = (xc * (1.0f / 112.0f) + 1.0f) * 0.5f * 223.0f;
        float x0f = floorf(gx), y0f = floorf(gy);
        int x0 = (int)x0f, y0 = (int)y0f;
        float wx1 = gx - x0f, wx0 = 1.0f - wx1;
        float wy1 = gy - y0f, wy0 = 1.0f - wy1;
        float vx0 = (x0 >= 0 && x0 <= 223) ? 1.0f : 0.0f;
        float vx1 = (x0 + 1 >= 0 && x0 + 1 <= 223) ? 1.0f : 0.0f;
        float vy0 = (y0 >= 0 && y0 <= 223) ? 1.0f : 0.0f;
        float vy1 = (y0 + 1 >= 0 && y0 + 1 <= 223) ? 1.0f : 0.0f;
        float w00 = wx0 * wy0 * vx0 * vy0, w10 = wx1 * wy0 * vx1 * vy0;
        float w01 = wx0 * wy1 * vx0 * vy1, w11 = wx1 * wy1 * vx1 * vy1;
        int xi0 = min(max(x0, 0), 223), xi1 = min(max(x0 + 1, 0), 223);
        int yi0 = min(max(y0, 0), 223), yi1 = min(max(y0 + 1, 0), 223);
        const float* xb = x + (size_t)b * 3 * IMGSZ * IMGSZ;
        #pragma unroll
        for (int c = 0; c < 3; ++c) {
            const float* img = xb + (size_t)c * IMGSZ * IMGSZ;
            float v = img[yi0 * IMGSZ + xi0] * w00 + img[yi0 * IMGSZ + xi1] * w10
                    + img[yi1 * IMGSZ + xi0] * w01 + img[yi1 * IMGSZ + xi1] * w11;
            Atile[spos * 72 + c * 20 + q] = f32_to_bf16_u(v);
        }
    }
    __syncthreads();

    f32x8 acc[6];
    #pragma unroll
    for (int j = 0; j < 6; ++j) {
        float bias = b1[j * 16 + (lane & 15)];
        #pragma unroll
        for (int v = 0; v < 8; ++v) acc[j][v] = bias;
    }
    #pragma unroll
    for (int kk = 0; kk < 2; ++kk) {
        bf16x16 af = load_frag_a(Atile + (16 * w) * 72 + kk * 32, lane, 72);
        #pragma unroll
        for (int j = 0; j < 6; ++j)
            acc[j] = __builtin_amdgcn_wmma_f32_16x16x32_bf16(
                false, af, false, bw[kk * 6 + j], (short)0, acc[j], false, false);
    }

    // epilogue: 4 consecutive M values share one patch -> packed b64 stores
    const int lhalf = lane >> 4;
    const int pro = ro / 5, krr = ro % 5;
    #pragma unroll
    for (int j = 0; j < 6; ++j) {
        int o = j * 16 + (lane & 15);
        #pragma unroll
        for (int h = 0; h < 2; ++h) {
            ushort4v pk;
            #pragma unroll
            for (int v2 = 0; v2 < 4; ++v2)
                pk[v2] = f32_to_bf16_u(acc[j][h * 4 + v2]);
            int ao  = ao0 + 16 * w + lhalf * 8 + h * 4;   // ka = 0..3 across v2
            int pao = ao >> 2;
            size_t idx = ((size_t)(b * NTOK + pro * 64 + pao)) * K2DIM
                       + o * 20 + krr * 4;
            *(ushort4v*)&T1[idx] = pk;
        }
    }
}

// ---------------------------------------------------------------------------
// K2: conv2 as GEMM  C[32768 x 96] = A[32768 x 1920] * B[1920 x 96] + b2
// 256 blocks x 128 threads; block M-tile 128; wave: 2 M-subtiles x 6 N-tiles.
// Double-buffered TDM (tensor_load_to_lds) pipeline; DMA applies the LDS row
// padding (64B data + 16B pad = 80B stride) the fragment loader expects.
// ---------------------------------------------------------------------------
#define LDK2 40   // LDS row stride in ushorts (80 B)

__global__ __launch_bounds__(128) void conv2_gemm_kernel(
    const unsigned short* __restrict__ T1, const unsigned short* __restrict__ W2t,
    const float* __restrict__ b2, float* __restrict__ out) {
    __shared__ unsigned short Alds[2][128 * LDK2];
    __shared__ unsigned short Blds[2][96 * LDK2];
    const int tid  = threadIdx.x;
    const int lane = tid & 31, w = tid >> 5;
    const int row0 = blockIdx.x * 128;
    const int NK   = K2DIM / 32;   // 60

    f32x8 acc[12];
    #pragma unroll
    for (int j = 0; j < 6; ++j) {
        float bias = b2[j * 16 + (lane & 15)];
        #pragma unroll
        for (int v = 0; v < 8; ++v) { acc[2 * j][v] = bias; acc[2 * j + 1][v] = bias; }
    }

#if HAVE_TDM
    if (tid < 32) {   // wave 0 drives the TDM (EXEC ignored; one issue per wave)
        tdm_load_tile_2d(T1 + (size_t)row0 * K2DIM, &Alds[0][0],
                         32, 128, K2DIM, (unsigned)(BATCH * NTOK), K2DIM);
        tdm_load_tile_2d(W2t, &Blds[0][0],
                         32, 96, K2DIM, EMBED, K2DIM);
    }
    for (int i = 0; i < NK; ++i) {
        const int cur = i & 1;
        if (tid < 32) {
            if (i + 1 < NK) {
                int k0n = (i + 1) * 32;
                tdm_load_tile_2d(T1 + (size_t)row0 * K2DIM + k0n, &Alds[cur ^ 1][0],
                                 32, 128, K2DIM, (unsigned)(BATCH * NTOK), K2DIM);
                tdm_load_tile_2d(W2t + k0n, &Blds[cur ^ 1][0],
                                 32, 96, K2DIM, EMBED, K2DIM);
                __builtin_amdgcn_s_wait_tensorcnt((short)2);  // tile i done
            } else {
                __builtin_amdgcn_s_wait_tensorcnt((short)0);
            }
        }
        __syncthreads();
        const unsigned short* Ab = &Alds[cur][0];
        const unsigned short* Bb = &Blds[cur][0];
        bf16x16 a0 = load_frag_a(Ab + (32 * w) * LDK2, lane, LDK2);
        bf16x16 a1 = load_frag_a(Ab + (32 * w + 16) * LDK2, lane, LDK2);
        #pragma unroll
        for (int j = 0; j < 6; ++j) {
            bf16x16 bf = load_frag_b(Bb + (j * 16) * LDK2, lane, LDK2);
            acc[2 * j] = __builtin_amdgcn_wmma_f32_16x16x32_bf16(
                false, a0, false, bf, (short)0, acc[2 * j], false, false);
            acc[2 * j + 1] = __builtin_amdgcn_wmma_f32_16x16x32_bf16(
                false, a1, false, bf, (short)0, acc[2 * j + 1], false, false);
        }
        __syncthreads();   // reads done before TDM refills this buffer
    }
#else
    // Fallback: synchronous VGPR->LDS staging (single buffer)
    for (int k0 = 0; k0 < K2DIM; k0 += 32) {
        {
            const unsigned short* src = T1 + (size_t)(row0 + tid) * K2DIM + k0;
            unsigned short* dst = &Alds[0][0] + tid * LDK2;
            ushort8 v0 = *(const ushort8*)(src);
            ushort8 v1 = *(const ushort8*)(src + 8);
            ushort8 v2 = *(const ushort8*)(src + 16);
            ushort8 v3 = *(const ushort8*)(src + 24);
            *(ushort8*)(dst)      = v0; *(ushort8*)(dst + 8)  = v1;
            *(ushort8*)(dst + 16) = v2; *(ushort8*)(dst + 24) = v3;
        }
        if (tid < 96) {
            const unsigned short* src = W2t + (size_t)tid * K2DIM + k0;
            unsigned short* dst = &Blds[0][0] + tid * LDK2;
            ushort8 v0 = *(const ushort8*)(src);
            ushort8 v1 = *(const ushort8*)(src + 8);
            ushort8 v2 = *(const ushort8*)(src + 16);
            ushort8 v3 = *(const ushort8*)(src + 24);
            *(ushort8*)(dst)      = v0; *(ushort8*)(dst + 8)  = v1;
            *(ushort8*)(dst + 16) = v2; *(ushort8*)(dst + 24) = v3;
        }
        __syncthreads();
        bf16x16 a0 = load_frag_a(&Alds[0][0] + (32 * w) * LDK2, lane, LDK2);
        bf16x16 a1 = load_frag_a(&Alds[0][0] + (32 * w + 16) * LDK2, lane, LDK2);
        #pragma unroll
        for (int j = 0; j < 6; ++j) {
            bf16x16 bf = load_frag_b(&Blds[0][0] + (j * 16) * LDK2, lane, LDK2);
            acc[2 * j] = __builtin_amdgcn_wmma_f32_16x16x32_bf16(
                false, a0, false, bf, (short)0, acc[2 * j], false, false);
            acc[2 * j + 1] = __builtin_amdgcn_wmma_f32_16x16x32_bf16(
                false, a1, false, bf, (short)0, acc[2 * j + 1], false, false);
        }
        __syncthreads();
    }
#endif

    const int lhalf = lane >> 4;
    #pragma unroll
    for (int j = 0; j < 6; ++j) {
        int o = j * 16 + (lane & 15);
        #pragma unroll
        for (int t = 0; t < 2; ++t) {
            f32x8 a = acc[2 * j + t];
            #pragma unroll
            for (int v = 0; v < 8; ++v) {
                int row = row0 + 32 * w + 16 * t + v + lhalf * 8;
                out[(size_t)row * EMBED + o] = a[v];
            }
        }
    }
}

// ---------------------------------------------------------------------------
extern "C" void kernel_launch(void* const* d_in, const int* in_sizes, int n_in,
                              void* d_out, int out_size, void* d_ws, size_t ws_size,
                              hipStream_t stream) {
    const float* x    = (const float*)d_in[0];
    const float* dist = (const float*)d_in[1];
    const float* w1   = (const float*)d_in[2];
    const float* b1   = (const float*)d_in[3];
    const float* w2   = (const float*)d_in[4];
    const float* b2   = (const float*)d_in[5];
    float* out = (float*)d_out;

    float*          r_table = (float*)((char*)d_ws + WS_RTAB);
    unsigned short* W1t     = (unsigned short*)((char*)d_ws + WS_W1T);
    unsigned short* W2t     = (unsigned short*)((char*)d_ws + WS_W2T);
    unsigned short* T1      = (unsigned short*)((char*)d_ws + WS_T1);
    float* theta_out = out + (size_t)BATCH * NTOK * EMBED;   // tail of d_out

    prep_kernel<<<dim3(128), dim3(256), 0, stream>>>(dist, w1, w2, r_table, W1t, W2t, theta_out);
    sample_conv1_kernel<<<dim3(BATCH * H1 * 4), dim3(128), 0, stream>>>(x, r_table, W1t, b1, T1);
    conv2_gemm_kernel<<<dim3((BATCH * NTOK) / 128), dim3(128), 0, stream>>>(T1, W2t, b2, out);
}
